// Net_3375844295202
// MI455X (gfx1250) — matrix operand order repaired and checked by reading, not compiled, not run on hardware
//
#include <hip/hip_runtime.h>

// ---------------------------------------------------------------------------
// PointTransformer block, MI455X (gfx1250, wave32).
//   Phase 1: node GEMMs (input MLP, lin_in, lin/src/dst heads) -> WMMA f16
//   Phase 2: per-edge pos-MLP + attention + online segment softmax + agg
//   Phase 3: output GEMM (w_out) + bn3 + skip + relu -> WMMA f16
// Edge structure is fixed: node i owns edges [i*16, i*16+16) plus self-loop.
// ---------------------------------------------------------------------------

typedef __attribute__((ext_vector_type(16))) _Float16 v16h;
typedef __attribute__((ext_vector_type(8)))  float    v8f;

#define DM   32
#define KNN  16
#define NEDG 17                       // KNN + self loop
static constexpr float BN_S = 0.99999500003749969f;   // 1/sqrt(1+1e-5)

// full-wave (32-lane) butterfly sum; every lane ends with the total
__device__ __forceinline__ float wsum32(float v) {
  v += __shfl_xor(v, 1, 32);
  v += __shfl_xor(v, 2, 32);
  v += __shfl_xor(v, 4, 32);
  v += __shfl_xor(v, 8, 32);
  v += __shfl_xor(v, 16, 32);
  return v;
}

// D = A(16x32,f16) x W(32x32,f32->f16) + bias, stored f32 to out[N][32].
// A-layout: lane holds row (lane&15), halves j -> K=(lane>>4)*16+j (same as B).
// B-layout: lane holds col (lane&15) of tile, halves j -> K=(lane>>4)*16+j.
// C/D layout: vgpr r -> (M=(lane>>4)*8+r, N=(lane&15)+tile*16).
__device__ __forceinline__ void head_gemm(const v16h a, const float* __restrict__ W,
                                          const float* __restrict__ bias,
                                          float* __restrict__ out,
                                          int node0, int bk, int bcol, int mrow) {
  v16h b0, b1;
#pragma unroll
  for (int j = 0; j < 16; ++j) {
    b0[j] = (_Float16)W[(bk + j) * DM + bcol];
    b1[j] = (_Float16)W[(bk + j) * DM + 16 + bcol];
  }
  v8f c0, c1;
  const float bi0 = bias[bcol], bi1 = bias[16 + bcol];
#pragma unroll
  for (int r = 0; r < 8; ++r) { c0[r] = bi0; c1[r] = bi1; }
  c0 = __builtin_amdgcn_wmma_f32_16x16x32_f16(false, a, false, b0, (short)0, c0, false, false);
  c1 = __builtin_amdgcn_wmma_f32_16x16x32_f16(false, a, false, b1, (short)0, c1, false, false);
#pragma unroll
  for (int r = 0; r < 8; ++r) {
    const int M = mrow + r;
    out[(node0 + M) * DM + bcol]      = c0[r];
    out[(node0 + M) * DM + 16 + bcol] = c1[r];
  }
}

// ---------------------------------------------------------------------------
// Phase 1: one wave = one 16-node tile. 8 waves / block.
// ---------------------------------------------------------------------------
__global__ __launch_bounds__(256) void pt_node_gemms(
    const float* __restrict__ x, const float* __restrict__ pos,
    const float* __restrict__ w_in, const float* __restrict__ bn_in_g, const float* __restrict__ bn_in_b,
    const float* __restrict__ w_lin_in, const float* __restrict__ bn1_g, const float* __restrict__ bn1_b,
    const float* __restrict__ w_lin, const float* __restrict__ b_lin,
    const float* __restrict__ w_src, const float* __restrict__ b_src,
    const float* __restrict__ w_dst, const float* __restrict__ b_dst,
    float* __restrict__ ws_h, float* __restrict__ ws_xl,
    float* __restrict__ ws_asrc, float* __restrict__ ws_adst, int n_tiles)
{
  __shared__ _Float16 lds_h1[8][16][DM];        // wave-private 16x32 f16 tiles
  const int lane = threadIdx.x & 31;
  const int wv   = threadIdx.x >> 5;
  const int tile = blockIdx.x * 8 + wv;
  if (tile >= n_tiles) return;                  // wave-uniform: EXEC stays all-1s
  const int node0 = tile * 16;
  const int row  = lane & 15;
  const int koff = (lane >> 4) * 8;             // unused: A-layout K base is bk
  const int bk   = (lane >> 4) * 16;            // B-layout K base
  const int bcol = lane & 15;                   // B/C/D column in tile
  const int mrow = (lane >> 4) * 8;             // C/D row base

  // --- input MLP: h = relu(bn(concat(pos,x) @ w_in)); built directly in A-layout
  float in6[6];
  in6[0] = pos[(node0 + row) * 3 + 0];
  in6[1] = pos[(node0 + row) * 3 + 1];
  in6[2] = pos[(node0 + row) * 3 + 2];
  in6[3] = x[(node0 + row) * 3 + 0];
  in6[4] = x[(node0 + row) * 3 + 1];
  in6[5] = x[(node0 + row) * 3 + 2];
  v16h a;
#pragma unroll
  for (int j = 0; j < 16; ++j) {
    const int c = bk + j;
    float d = 0.f;
#pragma unroll
    for (int k = 0; k < 6; ++k) d += in6[k] * w_in[k * DM + c];
    const float hv = fmaxf(bn_in_g[c] * (d * BN_S) + bn_in_b[c], 0.f);
    ws_h[(node0 + row) * DM + c] = hv;          // x_skip for phase 3
    a[j] = (_Float16)hv;
  }

  // --- h1 = relu(bn1(h @ w_lin_in)) : two 16x16x32 WMMAs
  v16h b0, b1;
#pragma unroll
  for (int j = 0; j < 16; ++j) {
    b0[j] = (_Float16)w_lin_in[(bk + j) * DM + bcol];
    b1[j] = (_Float16)w_lin_in[(bk + j) * DM + 16 + bcol];
  }
  v8f c0, c1;
#pragma unroll
  for (int r = 0; r < 8; ++r) { c0[r] = 0.f; c1[r] = 0.f; }
  c0 = __builtin_amdgcn_wmma_f32_16x16x32_f16(false, a, false, b0, (short)0, c0, false, false);
  c1 = __builtin_amdgcn_wmma_f32_16x16x32_f16(false, a, false, b1, (short)0, c1, false, false);

  // bn1 + relu, stash tile in LDS so it can be re-read in A-layout
#pragma unroll
  for (int r = 0; r < 8; ++r) {
    const int M = mrow + r;
    const float v0 = fmaxf(bn1_g[bcol]      * (c0[r] * BN_S) + bn1_b[bcol],      0.f);
    const float v1 = fmaxf(bn1_g[bcol + 16] * (c1[r] * BN_S) + bn1_b[bcol + 16], 0.f);
    lds_h1[wv][M][bcol]      = (_Float16)v0;
    lds_h1[wv][M][bcol + 16] = (_Float16)v1;
  }
  // same-wave LDS RAW: drain DS counter before the transposed re-read
  asm volatile("s_wait_dscnt 0" ::: "memory");
  v16h a2;
#pragma unroll
  for (int j = 0; j < 8; ++j) {
    a2[j]     = lds_h1[wv][row][bk + j];
    a2[8 + j] = lds_h1[wv][row][bk + 8 + j];
  }

  // --- three heads: xl = h1@w_lin+b, a_src = h1@w_src+b, a_dst = h1@w_dst+b
  head_gemm(a2, w_lin, b_lin, ws_xl,   node0, bk, bcol, mrow);
  head_gemm(a2, w_src, b_src, ws_asrc, node0, bk, bcol, mrow);
  head_gemm(a2, w_dst, b_dst, ws_adst, node0, bk, bcol, mrow);
}

// ---------------------------------------------------------------------------
// Phase 2: one wave = one node, lane = channel. 17-edge loop, online softmax.
// Gathers of a_src[j]/xl[j] are one contiguous 128B line per wave.
// ---------------------------------------------------------------------------
__global__ __launch_bounds__(256) void pt_edge_attn(
    const int* __restrict__ srcIdx, const float* __restrict__ pos,
    const float* __restrict__ ws_adst, const float* __restrict__ ws_asrc,
    const float* __restrict__ ws_xl,
    const float* __restrict__ w_p1, const float* __restrict__ b_p1,
    const float* __restrict__ bnp_g, const float* __restrict__ bnp_b,
    const float* __restrict__ w_p2, const float* __restrict__ b_p2,
    const float* __restrict__ bna1_g, const float* __restrict__ bna1_b,
    const float* __restrict__ w_a1, const float* __restrict__ b_a1,
    const float* __restrict__ bna2_g, const float* __restrict__ bna2_b,
    const float* __restrict__ w_a2, const float* __restrict__ b_a2,
    const float* __restrict__ bn2_g, const float* __restrict__ bn2_b,
    _Float16* __restrict__ ws_h2, int N)
{
  const int lane = threadIdx.x & 31;
  const int node = blockIdx.x * 8 + (threadIdx.x >> 5);
  if (node >= N) return;
  const int c  = lane;                 // feature channel
  const int da = c & 3;                // attention channel (share_planes=8 -> c%4)

  // destination-side values, live across all 17 edges
  const float pix = pos[node * 3 + 0], piy = pos[node * 3 + 1], piz = pos[node * 3 + 2];
  const float adst_c = ws_adst[node * DM + c];

  // per-channel constants
  const float wp2_0 = w_p2[0 * DM + c], wp2_1 = w_p2[1 * DM + c], wp2_2 = w_p2[2 * DM + c];
  const float bp2_c = b_p2[c];
  const float g1 = bna1_g[c] * BN_S, bb1 = bna1_b[c];
  const float wa1_0 = w_a1[c * 4 + 0], wa1_1 = w_a1[c * 4 + 1];
  const float wa1_2 = w_a1[c * 4 + 2], wa1_3 = w_a1[c * 4 + 3];
  // lane-uniform small tensors (L0-cached broadcasts)
  float wp1[9], bp1[3], gp[3], bpp[3];
#pragma unroll
  for (int k = 0; k < 9; ++k) wp1[k] = w_p1[k];
#pragma unroll
  for (int k = 0; k < 3; ++k) { bp1[k] = b_p1[k]; gp[k] = bnp_g[k] * BN_S; bpp[k] = bnp_b[k]; }
  const float ba1_0 = b_a1[0], ba1_1 = b_a1[1], ba1_2 = b_a1[2], ba1_3 = b_a1[3];
  float ga2[4], ba2[4], wa2c[4];
#pragma unroll
  for (int d = 0; d < 4; ++d) {
    ga2[d]  = bna2_g[d] * BN_S;
    ba2[d]  = bna2_b[d];
    wa2c[d] = w_a2[d * 4 + da];        // column of w_a2 this lane needs
  }
  const float ba2c = b_a2[da];

  float m = -INFINITY, ssum = 0.f, acc = 0.f;   // online softmax state
  for (int e = 0; e < NEDG; ++e) {
    const int j = (e < KNN) ? srcIdx[node * KNN + e] : node;   // self loop last
    const float dx = pix - pos[j * 3 + 0];
    const float dy = piy - pos[j * 3 + 1];
    const float dz = piz - pos[j * 3 + 2];
    // pos_nn layer 1 (3->3, lane-uniform)
    const float u0 = dx * wp1[0] + dy * wp1[3] + dz * wp1[6] + bp1[0];
    const float u1 = dx * wp1[1] + dy * wp1[4] + dz * wp1[7] + bp1[1];
    const float u2 = dx * wp1[2] + dy * wp1[5] + dz * wp1[8] + bp1[2];
    const float t0 = fmaxf(gp[0] * u0 + bpp[0], 0.f);
    const float t1 = fmaxf(gp[1] * u1 + bpp[1], 0.f);
    const float t2 = fmaxf(gp[2] * u2 + bpp[2], 0.f);
    // pos_nn layer 2 (3->32, per channel)
    const float delta_c = t0 * wp2_0 + t1 * wp2_1 + t2 * wp2_2 + bp2_c;
    // attention pre-activation
    float av = adst_c - ws_asrc[j * DM + c] + delta_c;
    av = fmaxf(g1 * av + bb1, 0.f);
    // 32 -> 4 projection: four wave reductions
    const float q0 = wsum32(av * wa1_0) + ba1_0;
    const float q1 = wsum32(av * wa1_1) + ba1_1;
    const float q2 = wsum32(av * wa1_2) + ba1_2;
    const float q3 = wsum32(av * wa1_3) + ba1_3;
    const float s0 = fmaxf(ga2[0] * q0 + ba2[0], 0.f);
    const float s1 = fmaxf(ga2[1] * q1 + ba2[1], 0.f);
    const float s2 = fmaxf(ga2[2] * q2 + ba2[2], 0.f);
    const float s3 = fmaxf(ga2[3] * q3 + ba2[3], 0.f);
    const float l  = s0 * wa2c[0] + s1 * wa2c[1] + s2 * wa2c[2] + s3 * wa2c[3] + ba2c;
    // grouped message for this channel
    const float msg = ws_xl[j * DM + c] + delta_c;
    // online softmax update (exact, single pass)
    const float mn = fmaxf(m, l);
    const float sc = __expf(m - mn);
    const float w  = __expf(l - mn);
    ssum = ssum * sc + w;
    acc  = acc * sc + w * msg;
    m = mn;
  }
  const float outv = acc / (ssum * (float)NEDG);               // softmax + mean
  const float h2 = fmaxf(bn2_g[c] * (outv * BN_S) + bn2_b[c], 0.f);
  ws_h2[node * DM + c] = (_Float16)h2;
}

// ---------------------------------------------------------------------------
// Phase 3: out = relu(bn3(h2 @ w_out) + x_skip). One wave = 16-node tile.
// ---------------------------------------------------------------------------
__global__ __launch_bounds__(256) void pt_out_gemm(
    const _Float16* __restrict__ ws_h2, const float* __restrict__ w_out,
    const float* __restrict__ bn3_g, const float* __restrict__ bn3_b,
    const float* __restrict__ ws_h, float* __restrict__ out, int n_tiles)
{
  const int lane = threadIdx.x & 31;
  const int tile = blockIdx.x * 8 + (threadIdx.x >> 5);
  if (tile >= n_tiles) return;
  const int node0 = tile * 16;
  const int row  = lane & 15;
  const int koff = (lane >> 4) * 8;
  const int bk   = (lane >> 4) * 16;
  const int bcol = lane & 15;
  const int mrow = (lane >> 4) * 8;

  v16h a, b0, b1;
#pragma unroll
  for (int j = 0; j < 8; ++j) {
    a[j]     = ws_h2[(node0 + row) * DM + bk + j];
    a[8 + j] = ws_h2[(node0 + row) * DM + bk + 8 + j];
  }
#pragma unroll
  for (int j = 0; j < 16; ++j) {
    b0[j] = (_Float16)w_out[(bk + j) * DM + bcol];
    b1[j] = (_Float16)w_out[(bk + j) * DM + 16 + bcol];
  }
  v8f c0, c1;
#pragma unroll
  for (int r = 0; r < 8; ++r) { c0[r] = 0.f; c1[r] = 0.f; }
  c0 = __builtin_amdgcn_wmma_f32_16x16x32_f16(false, a, false, b0, (short)0, c0, false, false);
  c1 = __builtin_amdgcn_wmma_f32_16x16x32_f16(false, a, false, b1, (short)0, c1, false, false);
#pragma unroll
  for (int r = 0; r < 8; ++r) {
    const int M = mrow + r;
    const float v0 = bn3_g[bcol]      * (c0[r] * BN_S) + bn3_b[bcol];
    const float v1 = bn3_g[bcol + 16] * (c1[r] * BN_S) + bn3_b[bcol + 16];
    out[(node0 + M) * DM + bcol]      = fmaxf(v0 + ws_h[(node0 + M) * DM + bcol],      0.f);
    out[(node0 + M) * DM + 16 + bcol] = fmaxf(v1 + ws_h[(node0 + M) * DM + 16 + bcol], 0.f);
  }
}

// ---------------------------------------------------------------------------
extern "C" void kernel_launch(void* const* d_in, const int* in_sizes, int n_in,
                              void* d_out, int out_size, void* d_ws, size_t ws_size,
                              hipStream_t stream) {
  // inputs: x, pos, edge_index, then params in dict insertion order
  const float* x        = (const float*)d_in[0];
  const float* pos      = (const float*)d_in[1];
  const int*   eidx     = (const int*)d_in[2];     // [2, N*K]; row 0 = src
  const float* w_in     = (const float*)d_in[3];
  const float* bn_in_g  = (const float*)d_in[4];
  const float* bn_in_b  = (const float*)d_in[5];
  const float* w_lin_in = (const float*)d_in[6];
  const float* w_out    = (const float*)d_in[7];
  const float* w_p1     = (const float*)d_in[8];
  const float* b_p1     = (const float*)d_in[9];
  const float* bnp_g    = (const float*)d_in[10];
  const float* bnp_b    = (const float*)d_in[11];
  const float* w_p2     = (const float*)d_in[12];
  const float* b_p2     = (const float*)d_in[13];
  const float* bna1_g   = (const float*)d_in[14];
  const float* bna1_b   = (const float*)d_in[15];
  const float* w_a1     = (const float*)d_in[16];
  const float* b_a1     = (const float*)d_in[17];
  const float* bna2_g   = (const float*)d_in[18];
  const float* bna2_b   = (const float*)d_in[19];
  const float* w_a2     = (const float*)d_in[20];
  const float* b_a2     = (const float*)d_in[21];
  const float* w_lin    = (const float*)d_in[22];
  const float* b_lin    = (const float*)d_in[23];
  const float* w_src    = (const float*)d_in[24];
  const float* b_src    = (const float*)d_in[25];
  const float* w_dst    = (const float*)d_in[26];
  const float* b_dst    = (const float*)d_in[27];
  const float* bn1_g    = (const float*)d_in[28];
  const float* bn1_b    = (const float*)d_in[29];
  const float* bn2_g    = (const float*)d_in[30];
  const float* bn2_b    = (const float*)d_in[31];
  const float* bn3_g    = (const float*)d_in[32];
  const float* bn3_b    = (const float*)d_in[33];

  const int N = in_sizes[0] / 3;         // 100000 (divisible by 16)
  const int n_tiles = N / 16;

  // workspace: h | xl | a_src | a_dst (f32 N*32 each) | h2 (f16 N*32)
  float* ws_h    = (float*)d_ws;
  float* ws_xl   = ws_h    + (size_t)N * DM;
  float* ws_asrc = ws_xl   + (size_t)N * DM;
  float* ws_adst = ws_asrc + (size_t)N * DM;
  _Float16* ws_h2 = (_Float16*)(ws_adst + (size_t)N * DM);

  dim3 blk(256);
  pt_node_gemms<<<(n_tiles + 7) / 8, blk, 0, stream>>>(
      x, pos, w_in, bn_in_g, bn_in_b, w_lin_in, bn1_g, bn1_b,
      w_lin, b_lin, w_src, b_src, w_dst, b_dst,
      ws_h, ws_xl, ws_asrc, ws_adst, n_tiles);

  pt_edge_attn<<<(N + 7) / 8, blk, 0, stream>>>(
      eidx, pos, ws_adst, ws_asrc, ws_xl,
      w_p1, b_p1, bnp_g, bnp_b, w_p2, b_p2,
      bna1_g, bna1_b, w_a1, b_a1, bna2_g, bna2_b, w_a2, b_a2,
      bn2_g, bn2_b, ws_h2, N);

  pt_out_gemm<<<(n_tiles + 7) / 8, blk, 0, stream>>>(
      ws_h2, w_out, bn3_g, bn3_b, ws_h, (float*)d_out, n_tiles);
}
